// SNN_72370198937916
// MI455X (gfx1250) — matrix operand order repaired and checked by reading, not compiled, not run on hardware
//
#include <hip/hip_runtime.h>

// ---------------------------------------------------------------------------
// SNN forward (LIF recurrent layer + LI readout), persistent-kernel design.
//
// 4 persistent workgroups (one per WGP), each owns a 128-column slice of the
// hidden state. w_rec slice (bf16, bank-conflict-free padded rows), v/i state
// (fp32, transposed + padded) are LDS-resident for all 1000 timesteps.
// Spikes z (bf16) are exchanged through L2 via a double-buffered global
// scratch buffer with an atomic grid barrier per step.
//
// Both GEMMs feeding i_in run on the matrix pipe (v_wmma_f32_16x16x32_bf16):
//   acc  = x_t(pad K=5->32) @ w_in.T      (1 wmma per N-tile, from LDS)
//   acc += z_prev @ w_rec_slice.T          (16 K-steps; all 8 B-tiles
//                                           preloaded per K-step for deep
//                                           ds/wmma pipelining)
// ---------------------------------------------------------------------------

typedef __attribute__((ext_vector_type(16))) __bf16 v16bf;
typedef __attribute__((ext_vector_type(8)))  __bf16 v8bf;
typedef __attribute__((ext_vector_type(8)))  float  v8f;
typedef __attribute__((ext_vector_type(4)))  float  v4f;

#define SEQ_LEN 1000
#define BATCH   128
#define IN_DIM  5
#define KPAD    32           // input GEMM K padded to one wmma K-step
#define HIDDEN  512
#define OUT_DIM 2

#define NBLK    4
#define SLICE   128          // hidden columns per block
#define THREADS 256          // 8 waves (wave32)

#define WSTRIDE 520          // w_rec LDS row stride (elems): bank stride 4
#define SSTRIDE 132          // v/i LDS row stride (floats): bank stride 4

// LDS layout (bytes)
#define WREC_BYTES (SLICE * WSTRIDE * 2)     // 133120  bf16 w_rec slice (padded)
#define V_BYTES    (SLICE * SSTRIDE * 4)     // 67584   membrane v [cl][b]
#define I_BYTES    (SLICE * SSTRIDE * 4)     // 67584   synaptic i [cl][b]
#define XP_BYTES   (BATCH * KPAD * 2)        // 8192    staged x_t, bf16, K-padded
#define WIP_BYTES  (SLICE * KPAD * 2)        // 8192    w_in slice, bf16, K-padded
#define WOUT_BYTES (OUT_DIM * HIDDEN * 4)    // 4096    w_out (block 0)
#define SMEM_BYTES (WREC_BYTES + V_BYTES + I_BYTES + XP_BYTES + WIP_BYTES + WOUT_BYTES)

// workspace layout: [0,512) barrier counter; then two z buffers (bf16)
#define Z_BYTES (BATCH * HIDDEN * 2)         // 131072 per buffer

__global__ void snn_init_ws(int* ws_i, int n_ints) {
  int i = blockIdx.x * blockDim.x + threadIdx.x;
  int stride = gridDim.x * blockDim.x;
  for (; i < n_ints; i += stride) ws_i[i] = 0;
}

__device__ __forceinline__ v16bf ld_b16x16(const __bf16* p) {
  v8bf a0 = *(const v8bf*)(p);
  v8bf a1 = *(const v8bf*)(p + 16);
  return __builtin_shufflevector(a0, a1,
           0,1,2,3,4,5,6,7,8,9,10,11,12,13,14,15);
}

__global__ void __launch_bounds__(THREADS, 1)
snn_persistent(const float* __restrict__ x,
               const float* __restrict__ w_in,
               const float* __restrict__ w_rec,
               const float* __restrict__ w_out,
               float* __restrict__ out,
               __bf16* __restrict__ zbuf0,
               __bf16* __restrict__ zbuf1,
               int* __restrict__ counter)
{
  extern __shared__ char smem[];
  __bf16* wrec_lds = (__bf16*)smem;
  float*  v_lds    = (float*)(smem + WREC_BYTES);
  float*  i_lds    = (float*)(smem + WREC_BYTES + V_BYTES);
  __bf16* xpad     = (__bf16*)(smem + WREC_BYTES + V_BYTES + I_BYTES);
  __bf16* winp     = (__bf16*)(smem + WREC_BYTES + V_BYTES + I_BYTES + XP_BYTES);
  float*  wout_lds = (float*)(smem + WREC_BYTES + V_BYTES + I_BYTES + XP_BYTES + WIP_BYTES);

  const int tid  = threadIdx.x;
  const int blk  = blockIdx.x;
  const int c0   = blk * SLICE;
  const int wv   = tid >> 5;        // wave id 0..7 (M-stripe)
  const int ln   = tid & 31;
  const int l15  = ln & 15;
  const int half = ln >> 4;         // lane half selects K chunk interleave

  // ---- one-time setup: weights -> LDS, state -> 0 ----
  for (int idx = tid; idx < SLICE * HIDDEN; idx += THREADS) {
    int r = idx >> 9;               // row within slice (hidden col c0+r)
    int k = idx & (HIDDEN - 1);
    wrec_lds[r * WSTRIDE + k] = (__bf16)w_rec[(c0 + r) * HIDDEN + k];
  }
  for (int idx = tid; idx < SLICE * SSTRIDE; idx += THREADS) {
    v_lds[idx] = 0.0f;
    i_lds[idx] = 0.0f;
  }
  for (int idx = tid; idx < SLICE * KPAD; idx += THREADS) {
    int r = idx >> 5;               // hidden col c0+r
    int k = idx & (KPAD - 1);
    winp[idx] = (k < IN_DIM) ? (__bf16)w_in[(c0 + r) * IN_DIM + k] : (__bf16)0.0f;
  }
  for (int idx = tid; idx < OUT_DIM * HIDDEN; idx += THREADS)
    wout_lds[idx] = w_out[idx];
  __syncthreads();

  // LI readout state (block 0 only; one element per thread: 128*2 == 256)
  float vo = 0.0f, io = 0.0f;
  const int b_out = tid >> 1;
  const int o_out = tid & 1;

  const int arow = (wv << 4) + l15;   // A-matrix batch row for this lane
  const int b0   = (wv << 4) + (half << 3);  // first batch row of C sub-block
  const int koA  = half ? 8  : 0;     // A K-chunk interleave (ISA layout)
  const int koB  = half ? 8  : 0;     // B K-chunk interleave (ISA layout)

  for (int t = 0; t < SEQ_LEN; ++t) {
    const __bf16* __restrict__ zprev = (t & 1) ? zbuf1 : zbuf0;
    __bf16* __restrict__ znext       = (t & 1) ? zbuf0 : zbuf1;

    // stage x_t into LDS as bf16, K-padded to 32 (zeros beyond IN_DIM)
    for (int idx = tid; idx < BATCH * KPAD; idx += THREADS) {
      int b = idx >> 5;
      int k = idx & (KPAD - 1);
      xpad[idx] = (k < IN_DIM) ? (__bf16)x[(t * BATCH + b) * IN_DIM + k]
                               : (__bf16)0.0f;
    }
    __syncthreads();

    // ---- seed accumulators with input GEMM: acc = x_t @ w_in_slice^T ----
    v8f acc[8];
    {
      const __bf16* ap = xpad + arow * KPAD + koA;
      v8bf a0 = *(const v8bf*)(ap);
      v8bf a1 = *(const v8bf*)(ap + 16);
      v16bf A = __builtin_shufflevector(a0, a1,
                 0,1,2,3,4,5,6,7,8,9,10,11,12,13,14,15);
      v16bf Bt[8];
#pragma unroll
      for (int tn = 0; tn < 8; ++tn)
        Bt[tn] = ld_b16x16(winp + (tn * 16 + l15) * KPAD + koB);
#pragma unroll
      for (int tn = 0; tn < 8; ++tn) {
        v8f zero;
#pragma unroll
        for (int q = 0; q < 8; ++q) zero[q] = 0.0f;
        acc[tn] = __builtin_amdgcn_wmma_f32_16x16x32_bf16(
            false, A, false, Bt[tn], (short)0, zero, false, false);
      }
    }

    // ---- recurrent GEMM: acc += z_prev @ w_rec_slice^T (M=128,N=128,K=512)
#pragma unroll 2
    for (int k0 = 0; k0 < HIDDEN; k0 += 32) {
      const __bf16* ap = zprev + arow * HIDDEN + k0 + koA;
      v8bf a0 = *(const v8bf*)(ap);
      v8bf a1 = *(const v8bf*)(ap + 16);
      v16bf A = __builtin_shufflevector(a0, a1,
                 0,1,2,3,4,5,6,7,8,9,10,11,12,13,14,15);
      __builtin_prefetch(zprev + arow * HIDDEN + k0 + 64, 0, 1);
      // phase 1: preload all 8 B tiles (16 outstanding ds_load_b128)
      v16bf Bt[8];
#pragma unroll
      for (int tn = 0; tn < 8; ++tn)
        Bt[tn] = ld_b16x16(wrec_lds + (tn * 16 + l15) * WSTRIDE + k0 + koB);
      // phase 2: drain into the matrix pipe
#pragma unroll
      for (int tn = 0; tn < 8; ++tn)
        acc[tn] = __builtin_amdgcn_wmma_f32_16x16x32_bf16(
            false, A, false, Bt[tn], (short)0, acc[tn], false, false);
    }

    // ---- LIF pointwise update + spike emission (vectorized state) ----
#pragma unroll
    for (int tn = 0; tn < 8; ++tn) {
      const int cl = tn * 16 + l15;           // local hidden column
      float* vp = v_lds + cl * SSTRIDE + b0;
      float* ip = i_lds + cl * SSTRIDE + b0;
      v4f vold0 = *(const v4f*)(vp);
      v4f vold1 = *(const v4f*)(vp + 4);
      v4f iold0 = *(const v4f*)(ip);
      v4f iold1 = *(const v4f*)(ip + 4);
      v4f vnew0, vnew1, inew0, inew1;
#pragma unroll
      for (int j = 0; j < 8; ++j) {
        const float vold = (j < 4) ? vold0[j & 3] : vold1[j & 3];
        const float iold = (j < 4) ? iold0[j & 3] : iold1[j & 3];
        const float vdec = 0.9f * vold + 0.1f * iold;   // leaky integration
        const float spike = (vdec > 1.0f);
        const float vnew = (vdec > 1.0f) ? 0.0f : vdec; // reset
        const float inew = 0.8f * iold + acc[tn][j];    // syn decay + i_in
        if (j < 4) { vnew0[j & 3] = vnew; inew0[j & 3] = inew; }
        else       { vnew1[j & 3] = vnew; inew1[j & 3] = inew; }
        znext[(b0 + j) * HIDDEN + c0 + cl] =
            (__bf16)((vdec > 1.0f) ? 1.0f : 0.0f);
        (void)spike;
      }
      *(v4f*)(vp)     = vnew0;
      *(v4f*)(vp + 4) = vnew1;
      *(v4f*)(ip)     = inew0;
      *(v4f*)(ip + 4) = inew1;
    }

    // ---- grid-wide barrier (4 resident blocks) ----
    __threadfence();
    __syncthreads();
    if (tid == 0) {
      __hip_atomic_fetch_add(counter, 1, __ATOMIC_ACQ_REL,
                             __HIP_MEMORY_SCOPE_AGENT);
      const int target = NBLK * (t + 1);
      while (__hip_atomic_load(counter, __ATOMIC_ACQUIRE,
                               __HIP_MEMORY_SCOPE_AGENT) < target) {
        __builtin_amdgcn_s_sleep(1);
      }
    }
    __syncthreads();

    // ---- LI readout (block 0 only; overlaps other blocks' next GEMM) ----
    if (blk == 0) {
      float y = 0.0f;
      const __bf16* zr = znext + b_out * HIDDEN;
#pragma unroll 2
      for (int h = 0; h < HIDDEN; h += 8) {
        v8bf zz = *(const v8bf*)(zr + h);
#pragma unroll
        for (int q = 0; q < 8; ++q)
          y += (float)zz[q] * wout_lds[o_out * HIDDEN + h + q];
      }
      const float vo_new = 0.9f * vo + 0.1f * io;  // uses old io
      io = 0.8f * io + y;
      vo = vo_new;
      out[t * (BATCH * OUT_DIM) + b_out * OUT_DIM + o_out] = vo;
    }
  }
}

extern "C" void kernel_launch(void* const* d_in, const int* in_sizes, int n_in,
                              void* d_out, int out_size, void* d_ws, size_t ws_size,
                              hipStream_t stream) {
  (void)in_sizes; (void)n_in; (void)out_size; (void)ws_size;

  const float* x     = (const float*)d_in[0];
  const float* w_in  = (const float*)d_in[1];
  const float* w_rec = (const float*)d_in[2];
  const float* w_out = (const float*)d_in[3];
  float* out = (float*)d_out;

  char* ws = (char*)d_ws;
  int*    counter = (int*)ws;
  __bf16* z0 = (__bf16*)(ws + 512);
  __bf16* z1 = (__bf16*)(ws + 512 + Z_BYTES);

  // zero counter + both z buffers (z0 must be zero: initial spikes are 0)
  const int n_ints = (512 + 2 * Z_BYTES) / 4;
  snn_init_ws<<<64, 256, 0, stream>>>((int*)ws, n_ints);

  hipFuncSetAttribute((const void*)snn_persistent,
                      hipFuncAttributeMaxDynamicSharedMemorySize, SMEM_BYTES);
  snn_persistent<<<NBLK, THREADS, SMEM_BYTES, stream>>>(
      x, w_in, w_rec, w_out, out, z0, z1, counter);
}
